// HypergraphPartitionModel_53661321396310
// MI455X (gfx1250) — matrix operand order repaired and checked by reading, not compiled, not run on hardware
//
#include <hip/hip_runtime.h>
#include <hip/hip_bf16.h>
#include <math.h>

#define NNODES 100000
#define NEDGES 50000
#define NNZV   800000
#define DHID   256
#define DLOW   64

typedef __attribute__((ext_vector_type(16))) __bf16          v16bf;
typedef __attribute__((ext_vector_type(2)))  __bf16          v2bf;
typedef __attribute__((ext_vector_type(8)))  float           v8f;
typedef __attribute__((ext_vector_type(8)))  unsigned short  ushort8;

union FragBF { v16bf bf; ushort8 us[2]; };

// native f32 -> bf16 (RNE on gfx1250), packed pair -> dword
__device__ __forceinline__ unsigned int pack2(float lo, float hi) {
  union { v2bf v; unsigned int u; } c;
  c.v.x = (__bf16)lo;
  c.v.y = (__bf16)hi;
  return c.u;
}

__device__ __forceinline__ unsigned short f2bf(float f) {
  union { __bf16 b; unsigned short u; } c;
  c.b = (__bf16)f;
  return c.u;
}

// ---------------------------------------------------------------- utility
__global__ void zero_kernel(float* __restrict__ p, long long n) {
  long long i = (long long)blockIdx.x * blockDim.x + threadIdx.x;
  long long stride = (long long)gridDim.x * blockDim.x;
  for (; i < n; i += stride) p[i] = 0.0f;
}

__global__ void degree_kernel(const int* __restrict__ ni, const int* __restrict__ ei,
                              float* __restrict__ D, float* __restrict__ B, int nnz) {
  int i = blockIdx.x * blockDim.x + threadIdx.x;
  if (i < nnz) {
    atomicAdd(&D[ni[i]], 1.0f);
    atomicAdd(&B[ei[i]], 1.0f);
  }
}

__global__ void invert_kernel(float* __restrict__ p, int n) {
  int i = blockIdx.x * blockDim.x + threadIdx.x;
  if (i < n) { float v = p[i]; p[i] = (v > 0.0f) ? (1.0f / v) : 0.0f; }
}

// ---------------------------------------------------------------- aligned WMMA GEMM
// Requires: K % 32 == 0, dout % 64 == 0, rows 16B-aligned (K % 4 == 0).
// 256 threads = 8 wave32. Block tile 128(M) x 64(N); wave tile 32x32
// -> 4 x v_wmma_f32_16x16x32_bf16 per k-step. Double-buffered LDS:
// one barrier per k-step; tile k+1 HBM fetch overlaps tile k WMMAs.
__global__ __launch_bounds__(256)
void gemm_wmma_aligned(const float* __restrict__ A, const float* __restrict__ B,
                       const float* __restrict__ bias, float* __restrict__ C,
                       int M, int K, int dout, int relu) {
  __shared__ unsigned short As[2][128 * 40];  // [row][k] bf16, stride 40 (bank-safe)
  __shared__ unsigned short Bs[2][64 * 40];   // [col][k] bf16 (transposed stage)

  const int tid   = threadIdx.x;
  const int lane  = tid & 31;
  const int wid   = tid >> 5;
  const int waveM = wid & 3;      // 4 groups of 32 rows
  const int waveN = wid >> 2;     // 2 groups of 32 cols
  const int lmod  = lane & 15;
  const int lhalf = lane >> 4;

  const int rb = blockIdx.x * 128;
  const int cb = blockIdx.y * 64;

  // staging coordinates
  const int ar  = tid >> 2;        // A row (chunk 0); chunk 1 adds 64
  const int akc = (tid & 3) * 8;   // A k-chunk
  const int bn  = tid & 63;        // B column slice (coalesced along n)
  const int bk  = (tid >> 6) * 8;  // B k-chunk

  auto stage = [&](int k0, int p) {
#pragma unroll
    for (int c = 0; c < 2; ++c) {
      int r = ar + c * 64;
      int gr = rb + r;
      uint4 pk = make_uint4(0u, 0u, 0u, 0u);
      if (gr < M) {
        const float* ap = A + (size_t)gr * K + (k0 + akc);
        float4 f0 = *(const float4*)ap;
        float4 f1 = *(const float4*)(ap + 4);
        pk.x = pack2(f0.x, f0.y); pk.y = pack2(f0.z, f0.w);
        pk.z = pack2(f1.x, f1.y); pk.w = pack2(f1.z, f1.w);
      }
      *(uint4*)&As[p][r * 40 + akc] = pk;
    }
    {
      const float* bp = B + (size_t)(k0 + bk) * dout + (cb + bn);
      float f0 = bp[0];
      float f1 = bp[(size_t)dout];
      float f2 = bp[(size_t)dout * 2];
      float f3 = bp[(size_t)dout * 3];
      float f4 = bp[(size_t)dout * 4];
      float f5 = bp[(size_t)dout * 5];
      float f6 = bp[(size_t)dout * 6];
      float f7 = bp[(size_t)dout * 7];
      uint4 pk;
      pk.x = pack2(f0, f1); pk.y = pack2(f2, f3);
      pk.z = pack2(f4, f5); pk.w = pack2(f6, f7);
      *(uint4*)&Bs[p][bn * 40 + bk] = pk;
    }
  };

  v8f acc00 = {}, acc01 = {}, acc10 = {}, acc11 = {};

  stage(0, 0);
  int p = 0;
  for (int k0 = 0; k0 < K; k0 += 32) {
    __syncthreads();   // buf[p] staged & prior readers of buf[p^1] done

    // ---- fragments from buf[p] (ISA 7.12.2 layouts)
    FragBF fa0, fa1, fb0, fb1;
    {
      const unsigned short* q0 = &As[p][(waveM * 32 + lmod) * 40 + lhalf * 8];
      fa0.us[0] = *(const ushort8*)q0;
      fa0.us[1] = *(const ushort8*)(q0 + 16);
      const unsigned short* q1 = &As[p][(waveM * 32 + 16 + lmod) * 40 + lhalf * 8];
      fa1.us[0] = *(const ushort8*)q1;
      fa1.us[1] = *(const ushort8*)(q1 + 16);
    }
    {
      const unsigned short* q0 = &Bs[p][(waveN * 32 + lmod) * 40 + lhalf * 8];
      fb0.us[0] = *(const ushort8*)q0;
      fb0.us[1] = *(const ushort8*)(q0 + 16);
      const unsigned short* q1 = &Bs[p][(waveN * 32 + 16 + lmod) * 40 + lhalf * 8];
      fb1.us[0] = *(const ushort8*)q1;
      fb1.us[1] = *(const ushort8*)(q1 + 16);
    }

    // ---- prefetch next tile into buf[p^1] while WMMAs run
    if (k0 + 32 < K) stage(k0 + 32, p ^ 1);

    acc00 = __builtin_amdgcn_wmma_f32_16x16x32_bf16(false, fa0.bf, false, fb0.bf,
                                                    (short)0, acc00, false, false);
    acc01 = __builtin_amdgcn_wmma_f32_16x16x32_bf16(false, fa0.bf, false, fb1.bf,
                                                    (short)0, acc01, false, false);
    acc10 = __builtin_amdgcn_wmma_f32_16x16x32_bf16(false, fa1.bf, false, fb0.bf,
                                                    (short)0, acc10, false, false);
    acc11 = __builtin_amdgcn_wmma_f32_16x16x32_bf16(false, fa1.bf, false, fb1.bf,
                                                    (short)0, acc11, false, false);
    p ^= 1;
  }

  // ---- epilogue: C/D layout -> VGPR r: M = r + 8*lhalf, N = lmod
  int n0 = cb + waveN * 32 + lmod;
  int n1 = n0 + 16;
  float b0 = bias ? bias[n0] : 0.0f;
  float b1 = bias ? bias[n1] : 0.0f;
#pragma unroll
  for (int r = 0; r < 8; ++r) {
    int m0 = rb + waveM * 32 + r + lhalf * 8;
    int m1 = m0 + 16;
    if (m0 < M) {
      float v0 = acc00[r] + b0, v1 = acc01[r] + b1;
      if (relu) { v0 = fmaxf(v0, 0.0f); v1 = fmaxf(v1, 0.0f); }
      C[(size_t)m0 * dout + n0] = v0;
      C[(size_t)m0 * dout + n1] = v1;
    }
    if (m1 < M) {
      float v0 = acc10[r] + b0, v1 = acc11[r] + b1;
      if (relu) { v0 = fmaxf(v0, 0.0f); v1 = fmaxf(v1, 0.0f); }
      C[(size_t)m1 * dout + n0] = v0;
      C[(size_t)m1 * dout + n1] = v1;
    }
  }
}

// ---------------------------------------------------------------- generic WMMA GEMM
// Handles ragged K (layer 1: K=5). 64x64 block tile, 8 waves, 2 WMMA/wave/k-step.
__global__ __launch_bounds__(256)
void gemm_wmma_generic(const float* __restrict__ A, const float* __restrict__ B,
                       const float* __restrict__ bias, float* __restrict__ C,
                       int M, int K, int dout, int relu) {
  __shared__ unsigned short As[64 * 40];
  __shared__ unsigned short Bs[64 * 40];

  const int tid   = threadIdx.x;
  const int lane  = tid & 31;
  const int wid   = tid >> 5;
  const int waveM = wid & 3;
  const int waveN = wid >> 2;
  const int lmod  = lane & 15;
  const int lhalf = lane >> 4;

  const int rb = blockIdx.x * 64;
  const int cb = blockIdx.y * 64;

  v8f acc0 = {};
  v8f acc1 = {};

  for (int k0 = 0; k0 < K; k0 += 32) {
    {
      int r = tid >> 2, c = (tid & 3) * 8;
      int gr = rb + r;
      const float* ap = A + (size_t)gr * K;
#pragma unroll
      for (int i = 0; i < 8; ++i) {
        int gk = k0 + c + i;
        float v = (gr < M && gk < K) ? ap[gk] : 0.0f;
        As[r * 40 + c + i] = f2bf(v);
      }
    }
#pragma unroll
    for (int i = 0; i < 8; ++i) {
      int idx = i * 256 + tid;
      int kk = idx >> 6, nn = idx & 63;
      int gk = k0 + kk;
      float v = (gk < K) ? B[(size_t)gk * dout + (cb + nn)] : 0.0f;
      Bs[nn * 40 + kk] = f2bf(v);
    }
    __syncthreads();

    FragBF fa, fb0, fb1;
    {
      const unsigned short* p = &As[(waveM * 16 + lmod) * 40 + lhalf * 8];
      fa.us[0] = *(const ushort8*)p;
      fa.us[1] = *(const ushort8*)(p + 16);
    }
    {
      const unsigned short* p = &Bs[(waveN * 32 + lmod) * 40 + lhalf * 8];
      fb0.us[0] = *(const ushort8*)p;
      fb0.us[1] = *(const ushort8*)(p + 16);
      const unsigned short* q = &Bs[(waveN * 32 + 16 + lmod) * 40 + lhalf * 8];
      fb1.us[0] = *(const ushort8*)q;
      fb1.us[1] = *(const ushort8*)(q + 16);
    }
    acc0 = __builtin_amdgcn_wmma_f32_16x16x32_bf16(false, fa.bf, false, fb0.bf,
                                                   (short)0, acc0, false, false);
    acc1 = __builtin_amdgcn_wmma_f32_16x16x32_bf16(false, fa.bf, false, fb1.bf,
                                                   (short)0, acc1, false, false);
    __syncthreads();
  }

  int n0 = cb + waveN * 32 + lmod;
  int n1 = n0 + 16;
  float b0 = bias ? bias[n0] : 0.0f;
  float b1 = bias ? bias[n1] : 0.0f;
#pragma unroll
  for (int r = 0; r < 8; ++r) {
    int mg = rb + waveM * 16 + r + lhalf * 8;
    if (mg < M) {
      float v0 = acc0[r] + b0;
      float v1 = acc1[r] + b1;
      if (relu) { v0 = fmaxf(v0, 0.0f); v1 = fmaxf(v1, 0.0f); }
      C[(size_t)mg * dout + n0] = v0;
      C[(size_t)mg * dout + n1] = v1;
    }
  }
}

// ---------------------------------------------------------------- scatters
__global__ __launch_bounds__(256)
void scatter_n2e(const float* __restrict__ xw, const int* __restrict__ ni,
                 const int* __restrict__ ei, float* __restrict__ eacc,
                 int nnz, int dout) {
  int tpe = dout >> 2;
  int epb = blockDim.x / tpe;
  int e_in_b = threadIdx.x / tpe;
  int c = (threadIdx.x % tpe) * 4;
  long long k = (long long)blockIdx.x * epb + e_in_b;
  if (k >= nnz) return;
  int node = ni[k], edge = ei[k];
  float4 v = *(const float4*)(xw + (size_t)node * dout + c);
  float* dst = eacc + (size_t)edge * dout + c;
  atomicAdd(dst + 0, v.x);
  atomicAdd(dst + 1, v.y);
  atomicAdd(dst + 2, v.z);
  atomicAdd(dst + 3, v.w);
}

__global__ __launch_bounds__(256)
void scatter_e2n(const float* __restrict__ eacc, const float* __restrict__ Binv,
                 const int* __restrict__ ni, const int* __restrict__ ei,
                 float* __restrict__ nacc, int nnz, int dout) {
  int tpe = dout >> 2;
  int epb = blockDim.x / tpe;
  int e_in_b = threadIdx.x / tpe;
  int c = (threadIdx.x % tpe) * 4;
  long long k = (long long)blockIdx.x * epb + e_in_b;
  if (k >= nnz) return;
  int node = ni[k], edge = ei[k];
  float s = Binv[edge];
  float4 v = *(const float4*)(eacc + (size_t)edge * dout + c);
  float* dst = nacc + (size_t)node * dout + c;
  atomicAdd(dst + 0, s * v.x);
  atomicAdd(dst + 1, s * v.y);
  atomicAdd(dst + 2, s * v.z);
  atomicAdd(dst + 3, s * v.w);
}

__global__ void finalize_elu(float* __restrict__ h, const float* __restrict__ Dinv,
                             const float* __restrict__ bias, int n, int dout) {
  long long i = (long long)blockIdx.x * blockDim.x + threadIdx.x;
  if (i < (long long)n * dout) {
    int row = (int)(i / dout), j = (int)(i % dout);
    float v = Dinv[row] * h[i] + bias[j];
    h[i] = (v > 0.0f) ? v : (__expf(v) - 1.0f);
  }
}

__global__ void head_softmax(const float* __restrict__ h, const float* __restrict__ fow,
                             const float* __restrict__ fob, float* __restrict__ out, int n) {
  int i = blockIdx.x * blockDim.x + threadIdx.x;
  if (i >= n) return;
  const float* row = h + (size_t)i * DLOW;
  float l0 = fob[0], l1 = fob[1];
#pragma unroll
  for (int j = 0; j < DLOW; ++j) {
    float v = row[j];
    l0 += v * fow[j * 2 + 0];
    l1 += v * fow[j * 2 + 1];
  }
  float m  = fmaxf(l0, l1);
  float e0 = __expf(l0 - m), e1 = __expf(l1 - m);
  float s  = e0 + e1;
  out[i * 2 + 0] = e0 / s;
  out[i * 2 + 1] = e1 / s;
}

// ---------------------------------------------------------------- launch
extern "C" void kernel_launch(void* const* d_in, const int* in_sizes, int n_in,
                              void* d_out, int out_size, void* d_ws, size_t ws_size,
                              hipStream_t stream) {
  (void)in_sizes; (void)n_in; (void)out_size; (void)ws_size;

  const float* x    = (const float*)d_in[0];
  const int*   hidx = (const int*)d_in[1];
  const int* ni = hidx;            // [NNZ] node indices
  const int* ei = hidx + NNZV;     // [NNZ] edge indices
  const float* W1 = (const float*)d_in[2];  const float* b1 = (const float*)d_in[3];
  const float* W2 = (const float*)d_in[4];  const float* b2 = (const float*)d_in[5];
  const float* W3 = (const float*)d_in[6];  const float* b3 = (const float*)d_in[7];
  const float* W4 = (const float*)d_in[8];  const float* b4 = (const float*)d_in[9];
  const float* W5 = (const float*)d_in[10]; const float* b5 = (const float*)d_in[11];
  const float* fw1 = (const float*)d_in[12]; const float* fb1 = (const float*)d_in[13];
  const float* fw2 = (const float*)d_in[14]; const float* fb2 = (const float*)d_in[15];
  const float* fow = (const float*)d_in[16]; const float* fob = (const float*)d_in[17];
  float* out = (float*)d_out;

  float* ws = (float*)d_ws;
  size_t off = 0;
  float* Dinv = ws + off; off += (size_t)((NNODES + 255) & ~255);
  float* Binv = ws + off; off += (size_t)((NEDGES + 255) & ~255);
  float* bufA = ws + off; off += (size_t)NNODES * DHID;   // GEMM output (xw)
  float* bufB = ws + off; off += (size_t)NNODES * DHID;   // layer in/out
  float* eacc = ws + off; off += (size_t)NEDGES * DHID;   // hyperedge accumulator

  const int T = 256;

  // degrees -> inverse degrees (recomputed each call: deterministic)
  zero_kernel<<<1024, T, 0, stream>>>(Dinv, NNODES);
  zero_kernel<<<1024, T, 0, stream>>>(Binv, NEDGES);
  degree_kernel<<<(NNZV + T - 1) / T, T, 0, stream>>>(ni, ei, Dinv, Binv, NNZV);
  invert_kernel<<<(NNODES + T - 1) / T, T, 0, stream>>>(Dinv, NNODES);
  invert_kernel<<<(NEDGES + T - 1) / T, T, 0, stream>>>(Binv, NEDGES);

  auto gemm = [&](const float* A, const float* W, const float* bias, float* C,
                  int K, int dout, int relu) {
    if ((K & 31) == 0) {
      dim3 g((NNODES + 127) / 128, dout / 64);
      gemm_wmma_aligned<<<g, T, 0, stream>>>(A, W, bias, C, NNODES, K, dout, relu);
    } else {
      dim3 g((NNODES + 63) / 64, dout / 64);
      gemm_wmma_generic<<<g, T, 0, stream>>>(A, W, bias, C, NNODES, K, dout, relu);
    }
  };

  auto hconv = [&](const float* hin, const float* W, const float* b, int K, int dout) {
    gemm(hin, W, nullptr, bufA, K, dout, 0);
    zero_kernel<<<2048, T, 0, stream>>>(eacc, (long long)NEDGES * dout);
    int epb = T / (dout >> 2);
    int sgrid = (NNZV + epb - 1) / epb;
    scatter_n2e<<<sgrid, T, 0, stream>>>(bufA, ni, ei, eacc, NNZV, dout);
    zero_kernel<<<2048, T, 0, stream>>>(bufB, (long long)NNODES * dout);
    scatter_e2n<<<sgrid, T, 0, stream>>>(eacc, Binv, ni, ei, bufB, NNZV, dout);
    long long tot = (long long)NNODES * dout;
    finalize_elu<<<(int)((tot + T - 1) / T), T, 0, stream>>>(bufB, Dinv, b, NNODES, dout);
  };

  hconv(x,    W1, b1, 5,    DHID);   // 5   -> 256 (generic path)
  hconv(bufB, W2, b2, DHID, DHID);   // 256 -> 256 (aligned path)
  hconv(bufB, W3, b3, DHID, DHID);
  hconv(bufB, W4, b4, DHID, DHID);
  hconv(bufB, W5, b5, DHID, DLOW);   // 256 -> 64

  // MLP head (bias + relu fused in GEMM epilogue)
  gemm(bufB, fw1, fb1, bufA, DLOW, DLOW, 1);
  gemm(bufA, fw2, fb2, bufB, DLOW, DLOW, 1);

  head_softmax<<<(NNODES + T - 1) / T, T, 0, stream>>>(bufB, fow, fob, out, NNODES);
}
